// IPABlock_1623497638537
// MI455X (gfx1250) — matrix operand order repaired and checked by reading, not compiled, not run on hardware
//
#include <hip/hip_runtime.h>
#include <cstddef>

// ---------------------------------------------------------------------------
// Types / constants
// ---------------------------------------------------------------------------
typedef __attribute__((ext_vector_type(16))) __bf16 v16bf;
typedef __attribute__((ext_vector_type(8)))  float  v8f;

#define N_TOK   512
#define DIM_C   512
#define H_C     8
#define CATW    1280   // H*(DV + 3*PV + PV + DPAIR) = 8*(16+12+4+128)

#define SCALAR_SCALE 0.14433756729740643f   // (3*16)^-0.5
#define POINT_SCALE  0.13608276348795434f   // (3*4*4.5)^-0.5
#define PAIR_SCALE   0.57735026918962576f   // 3^-0.5
#define EPS_C   1e-8f
#define LN_EPS_C 1e-3f

// WMMA bf16 fragment index helpers (per CDNA5 ISA §7.12.2, 16-bit layouts).
// A (16x32): lane = M | (grp<<4); element e<8 -> K = 8*grp+e ; e>=8 -> K = 8+8*grp+e
__device__ __forceinline__ int a_kidx(int grp, int e) {
  return (e < 8) ? (8 * grp + e) : (8 + 8 * grp + e);
}
// B (32x16): lane = N | (grp<<4); element e<8 -> K = 8*grp+e ; e>=8 -> K = 8+8*grp+e
__device__ __forceinline__ int b_kidx(int grp, int e) { return (e < 8) ? (8 * grp + e) : (8 + 8 * grp + e); }
// C/D (16x16 f32): lane = N | (grp<<4); VGPR r -> M = r + 8*grp

__device__ __forceinline__ v8f wmma_bf16(v16bf a, v16bf b, v8f c) {
  return __builtin_amdgcn_wmma_f32_16x16x32_bf16(false, a, false, b, (short)0, c,
                                                 false, false);
}

__device__ __forceinline__ float block_reduce_sum(float x, float* red) {
  int t = threadIdx.x;
  red[t] = x; __syncthreads();
  for (int s = 64; s > 0; s >>= 1) { if (t < s) red[t] += red[t + s]; __syncthreads(); }
  float r = red[0]; __syncthreads();
  return r;
}
__device__ __forceinline__ float block_reduce_max(float x, float* red) {
  int t = threadIdx.x;
  red[t] = x; __syncthreads();
  for (int s = 64; s > 0; s >>= 1) { if (t < s) red[t] = fmaxf(red[t], red[t + s]); __syncthreads(); }
  float r = red[0]; __syncthreads();
  return r;
}

// ---------------------------------------------------------------------------
// Generic GEMM: C[M,N] = A[M,K] @ B[K,N] (+bias)(+relu), fp32 io, bf16 WMMA.
// Block = 128 thr (4 waves). Block tile 64x32, K-step 32. Requires M%64==0,
// N%32==0, K%32==0 (all call sites satisfy this -> EXEC always all-ones).
// ---------------------------------------------------------------------------
__global__ void __launch_bounds__(128)
gemm_bf16_kernel(const float* __restrict__ A, const float* __restrict__ Bw,
                 const float* __restrict__ bias, float* __restrict__ C,
                 int M, int Nn, int K, int relu)
{
  __shared__ __bf16 As[64][32];
  __shared__ __bf16 Bs[32][32];
  const int tid  = threadIdx.x;
  const int wave = tid >> 5;
  const int lane = tid & 31;
  const int grp  = lane >> 4;
  const int ln16 = lane & 15;
  const int row0 = blockIdx.x * 64;
  const int col0 = blockIdx.y * 32;
  (void)M;

  v8f acc0 = {}; v8f acc1 = {};

  for (int k0 = 0; k0 < K; k0 += 32) {
    { // A tile 64x32 (coalesced b128 loads, cvt to bf16, ds_store)
      int r = tid >> 1, half = tid & 1;
      const float* src = A + (size_t)(row0 + r) * K + k0 + half * 16;
      #pragma unroll
      for (int v = 0; v < 4; ++v) {
        float4 f = reinterpret_cast<const float4*>(src)[v];
        int c = half * 16 + v * 4;
        As[r][c+0] = (__bf16)f.x; As[r][c+1] = (__bf16)f.y;
        As[r][c+2] = (__bf16)f.z; As[r][c+3] = (__bf16)f.w;
      }
    }
    { // B tile 32x32
      int kr = tid >> 2, q = tid & 3;
      const float* src = Bw + (size_t)(k0 + kr) * Nn + col0 + q * 8;
      #pragma unroll
      for (int v = 0; v < 2; ++v) {
        float4 f = reinterpret_cast<const float4*>(src)[v];
        int c = q * 8 + v * 4;
        Bs[kr][c+0] = (__bf16)f.x; Bs[kr][c+1] = (__bf16)f.y;
        Bs[kr][c+2] = (__bf16)f.z; Bs[kr][c+3] = (__bf16)f.w;
      }
    }
    __syncthreads();
    v16bf af, bf0, bf1;
    const int m = wave * 16 + ln16;
    #pragma unroll
    for (int e = 0; e < 16; ++e) af[e] = As[m][a_kidx(grp, e)];
    #pragma unroll
    for (int e = 0; e < 16; ++e) {
      int kk = b_kidx(grp, e);
      bf0[e] = Bs[kk][ln16];
      bf1[e] = Bs[kk][16 + ln16];
    }
    acc0 = wmma_bf16(af, bf0, acc0);
    acc1 = wmma_bf16(af, bf1, acc1);
    __syncthreads();
  }

  #pragma unroll
  for (int r = 0; r < 8; ++r) {
    int mrow = row0 + wave * 16 + r + 8 * grp;
    int nc   = col0 + ln16;
    float v0 = acc0[r], v1 = acc1[r];
    if (bias) { v0 += bias[nc]; v1 += bias[nc + 16]; }
    if (relu) { v0 = fmaxf(v0, 0.f); v1 = fmaxf(v1, 0.f); }
    C[(size_t)mrow * Nn + nc]      = v0;
    C[(size_t)mrow * Nn + nc + 16] = v1;
  }
}

// ---------------------------------------------------------------------------
// Transform qp/kp/vp (local, already in buffers) to global frame, in place.
// One thread per (tensor, b, n, h, p).  3*B*N*H*4 = 98304 threads.
// out[r] = sum_c pt[c]*R[c,r] + t[r]
// ---------------------------------------------------------------------------
__global__ void __launch_bounds__(256)
point_global_kernel(float* __restrict__ qp, float* __restrict__ kp,
                    float* __restrict__ vp, const float* __restrict__ rot,
                    const float* __restrict__ trans)
{
  int t = blockIdx.x * 256 + threadIdx.x;
  int which = t / 32768, rem = t % 32768;
  int p = rem & 3, h = (rem >> 2) & 7, n = (rem >> 5) & 511, b = rem >> 14;
  float* buf = (which == 0) ? qp : (which == 1) ? kp : vp;
  size_t off = (((size_t)b * N_TOK + n) * H_C + h) * 12 + p * 3;
  float c0 = buf[off], c1 = buf[off + 1], c2 = buf[off + 2];
  const float* R = rot   + ((size_t)b * N_TOK + n) * 9;
  const float* T = trans + ((size_t)b * N_TOK + n) * 3;
  #pragma unroll
  for (int r = 0; r < 3; ++r)
    buf[off + r] = c0 * R[r] + c1 * R[3 + r] + c2 * R[6 + r] + T[r];
}

// q2w/k2w: w_h * sum_p |pt|^2 per (b,h,n).  2*B*H*N = 16384 threads.
__global__ void __launch_bounds__(256)
sq_kernel(const float* __restrict__ qp, const float* __restrict__ kp,
          const float* __restrict__ pw, float* __restrict__ q2w,
          float* __restrict__ k2w)
{
  int t = blockIdx.x * 256 + threadIdx.x;
  int which = t >> 13, rem = t & 8191;
  int n = rem & 511, h = (rem >> 9) & 7, b = rem >> 12;
  const float* buf = which ? kp : qp;
  size_t off = (((size_t)b * N_TOK + n) * H_C + h) * 12;
  float s = 0.f;
  #pragma unroll
  for (int e = 0; e < 12; ++e) { float v = buf[off + e]; s += v * v; }
  float w = 0.5f * log1pf(expf(pw[h])) * POINT_SCALE;
  (which ? k2w : q2w)[((size_t)b * H_C + h) * N_TOK + n] = s * w;
}

// ---------------------------------------------------------------------------
// Pair logits: S[b,h,i,j] = (pairwise[b,i,j,:] @ Wpair + b_pair[h])*PAIR_SCALE
// One wave per (b, i, 16-wide j tile): A = 16 pairwise rows, K-loop over 128,
// B = Wpair (N padded 8->16 with zeros).  First streaming pass over pairwise.
// ---------------------------------------------------------------------------
__global__ void __launch_bounds__(128)
pair_logits_kernel(const float* __restrict__ pair, const float* __restrict__ Wpair,
                   const float* __restrict__ b_pair, float* __restrict__ S)
{
  int wid  = blockIdx.x * 4 + (threadIdx.x >> 5);
  int lane = threadIdx.x & 31, grp = lane >> 4, ln16 = lane & 15;
  int jt = wid & 31, i = (wid >> 5) & 511, b = wid >> 14;
  int j0 = jt * 16;

  v16bf bw[4];
  #pragma unroll
  for (int s = 0; s < 4; ++s)
    #pragma unroll
    for (int e = 0; e < 16; ++e) {
      int k = s * 32 + b_kidx(grp, e);
      bw[s][e] = (ln16 < 8) ? (__bf16)Wpair[k * 8 + ln16] : (__bf16)0.f;
    }

  const float* prow = pair + (((size_t)b * N_TOK + i) * N_TOK + j0 + ln16) * 128;
  __builtin_prefetch(prow, 0, 1);
  v8f acc = {};
  #pragma unroll
  for (int s = 0; s < 4; ++s) {
    v16bf af;
    #pragma unroll
    for (int e = 0; e < 16; ++e) af[e] = (__bf16)prow[s * 32 + a_kidx(grp, e)];
    acc = wmma_bf16(af, bw[s], acc);
  }
  if (ln16 < 8) {  // lane = head, VGPR r -> j = j0 + r + 8*grp (8 consecutive)
    float bp = b_pair[ln16];
    float* dst = S + (((size_t)b * H_C + ln16) * N_TOK + i) * N_TOK + j0 + 8 * grp;
    #pragma unroll
    for (int r = 0; r < 8; ++r) dst[r] = (acc[r] + bp) * PAIR_SCALE;
  }
}

// ---------------------------------------------------------------------------
// Attention logits: S += SCALAR_SCALE*(qs.ks) + 2*w_h*(qp.kp) - q2w_i - k2w_j
// One wave per (b,h,16x16 tile).  K=16 / K=12 zero-padded to 32.
// ---------------------------------------------------------------------------
__global__ void __launch_bounds__(128)
attn_logits_kernel(const float* __restrict__ qs, const float* __restrict__ ks,
                   const float* __restrict__ qp, const float* __restrict__ kp,
                   const float* __restrict__ q2w, const float* __restrict__ k2w,
                   const float* __restrict__ pw, float* __restrict__ S)
{
  int wid  = blockIdx.x * 4 + (threadIdx.x >> 5);
  int lane = threadIdx.x & 31, grp = lane >> 4, ln16 = lane & 15;
  int jt = wid & 31, it = (wid >> 5) & 31, h = (wid >> 10) & 7, b = wid >> 13;
  int i0 = it * 16, j0 = jt * 16;

  v16bf aq, bk, ap, bp;
  const float* qrow  = qs + (((size_t)b * N_TOK + i0 + ln16) * H_C + h) * 16;
  const float* krow  = ks + (((size_t)b * N_TOK + j0 + ln16) * H_C + h) * 16;
  const float* qprow = qp + (((size_t)b * N_TOK + i0 + ln16) * H_C + h) * 12;
  const float* kprow = kp + (((size_t)b * N_TOK + j0 + ln16) * H_C + h) * 12;
  #pragma unroll
  for (int e = 0; e < 16; ++e) {
    int ka = a_kidx(grp, e);
    int kb = b_kidx(grp, e);
    aq[e] = (ka < 16) ? (__bf16)qrow[ka]  : (__bf16)0.f;
    bk[e] = (kb < 16) ? (__bf16)krow[kb]  : (__bf16)0.f;
    ap[e] = (ka < 12) ? (__bf16)qprow[ka] : (__bf16)0.f;
    bp[e] = (kb < 12) ? (__bf16)kprow[kb] : (__bf16)0.f;
  }
  v8f ss = {}, pt = {};
  ss = wmma_bf16(aq, bk, ss);
  pt = wmma_bf16(ap, bp, pt);

  float w  = 0.5f * log1pf(expf(pw[h])) * POINT_SCALE;
  float k2 = k2w[((size_t)b * H_C + h) * N_TOK + j0 + ln16];
  #pragma unroll
  for (int r = 0; r < 8; ++r) {
    int i = i0 + r + 8 * grp;
    float q2 = q2w[((size_t)b * H_C + h) * N_TOK + i];
    size_t idx = (((size_t)b * H_C + h) * N_TOK + i) * N_TOK + j0 + ln16;
    S[idx] += SCALAR_SCALE * ss[r] + 2.f * w * pt[r] - q2 - k2;
  }
}

// Row softmax over j (in place).  One 128-thread block per (b,h,i) row.
__global__ void __launch_bounds__(128)
softmax_kernel(float* __restrict__ S)
{
  __shared__ float red[128];
  float* row = S + (size_t)blockIdx.x * N_TOK;
  int t = threadIdx.x;
  float v[4], m = -3.4e38f;
  #pragma unroll
  for (int k = 0; k < 4; ++k) { v[k] = row[t + 128 * k]; m = fmaxf(m, v[k]); }
  m = block_reduce_max(m, red);
  float s = 0.f;
  #pragma unroll
  for (int k = 0; k < 4; ++k) { v[k] = expf(v[k] - m); s += v[k]; }
  s = block_reduce_sum(s, red);
  float inv = 1.f / s;
  #pragma unroll
  for (int k = 0; k < 4; ++k) row[t + 128 * k] = v[k] * inv;
}

// ---------------------------------------------------------------------------
// out_s = attn @ vs (16 ch), out_pg = attn @ vp (12 ch).  One wave per
// (b,h,16-i tile), K-loop over all 512 j.  out_s written into cat[0:128].
// ---------------------------------------------------------------------------
__global__ void __launch_bounds__(128)
av_kernel(const float* __restrict__ S, const float* __restrict__ vs,
          const float* __restrict__ vp, float* __restrict__ cat,
          float* __restrict__ opg)
{
  int wid  = blockIdx.x * 4 + (threadIdx.x >> 5);
  int lane = threadIdx.x & 31, grp = lane >> 4, ln16 = lane & 15;
  int it = wid & 31, h = (wid >> 5) & 7, b = wid >> 8;
  int i0 = it * 16;

  v8f accs = {}, accp = {};
  for (int j0 = 0; j0 < N_TOK; j0 += 32) {
    v16bf aa, bvs, bvp;
    const float* arow = S + (((size_t)b * H_C + h) * N_TOK + i0 + ln16) * N_TOK + j0;
    #pragma unroll
    for (int e = 0; e < 16; ++e) aa[e] = (__bf16)arow[a_kidx(grp, e)];
    #pragma unroll
    for (int e = 0; e < 16; ++e) {
      int j = j0 + b_kidx(grp, e);
      bvs[e] = (__bf16)vs[(((size_t)b * N_TOK + j) * H_C + h) * 16 + ln16];
      bvp[e] = (ln16 < 12) ? (__bf16)vp[(((size_t)b * N_TOK + j) * H_C + h) * 12 + ln16]
                           : (__bf16)0.f;
    }
    accs = wmma_bf16(aa, bvs, accs);
    accp = wmma_bf16(aa, bvp, accp);
  }
  #pragma unroll
  for (int r = 0; r < 8; ++r) {
    int i = i0 + r + 8 * grp;
    cat[((size_t)b * N_TOK + i) * CATW + h * 16 + ln16] = accs[r];
    if (ln16 < 12)
      opg[(((size_t)b * N_TOK + i) * H_C + h) * 12 + ln16] = accp[r];
  }
}

// Local-frame back-transform + norms.  One thread per (b,n,h,p).
// cat[128:224] = out_pl, cat[224:256] = out_pn.
__global__ void __launch_bounds__(256)
pl_kernel(const float* __restrict__ opg, const float* __restrict__ rot,
          const float* __restrict__ trans, float* __restrict__ cat)
{
  int t = blockIdx.x * 256 + threadIdx.x;  // 32768
  int p = t & 3, h = (t >> 2) & 7, n = (t >> 5) & 511, b = t >> 14;
  size_t off = (((size_t)b * N_TOK + n) * H_C + h) * 12 + p * 3;
  const float* R = rot   + ((size_t)b * N_TOK + n) * 9;
  const float* T = trans + ((size_t)b * N_TOK + n) * 3;
  float g0 = opg[off] - T[0], g1 = opg[off + 1] - T[1], g2 = opg[off + 2] - T[2];
  size_t base = ((size_t)b * N_TOK + n) * CATW;
  float nrm = 0.f;
  #pragma unroll
  for (int c = 0; c < 3; ++c) {
    float l = g0 * R[c * 3 + 0] + g1 * R[c * 3 + 1] + g2 * R[c * 3 + 2];
    cat[base + 128 + (h * 4 + p) * 3 + c] = l;
    nrm += l * l;
  }
  cat[base + 224 + h * 4 + p] = sqrtf(nrm + EPS_C);
}

// ---------------------------------------------------------------------------
// out_pair[b,i,h,d] = sum_j attn[b,h,i,j] * pairwise[b,i,j,d]
// Block per (b,i): GEMM M=8(h, padded 16) x N=128(d) x K=512(j).
// Cooperative 32x128 fp32->bf16 tile in LDS; 4 waves x 2 d-tiles.
// Second (dominant) streaming pass over pairwise -> prefetch next tile.
// ---------------------------------------------------------------------------
__global__ void __launch_bounds__(128)
out_pair_kernel(const float* __restrict__ S, const float* __restrict__ pair,
                float* __restrict__ cat)
{
  __shared__ __bf16 Ps[32][128];
  int b = blockIdx.x >> 9, i = blockIdx.x & 511;
  int tid = threadIdx.x, wave = tid >> 5, lane = tid & 31;
  int grp = lane >> 4, ln16 = lane & 15;

  v8f acc[2] = {{}, {}};
  const float* pbase = pair + ((size_t)b * N_TOK + i) * N_TOK * 128;
  const int jr = tid >> 2, q = tid & 3;

  for (int j0 = 0; j0 < N_TOK; j0 += 32) {
    { // stage pairwise tile (coalesced b128 loads -> bf16 ds stores)
      const float* src = pbase + (size_t)(j0 + jr) * 128 + q * 32;
      #pragma unroll
      for (int v = 0; v < 8; ++v) {
        float4 f = reinterpret_cast<const float4*>(src)[v];
        int c = q * 32 + v * 4;
        Ps[jr][c+0] = (__bf16)f.x; Ps[jr][c+1] = (__bf16)f.y;
        Ps[jr][c+2] = (__bf16)f.z; Ps[jr][c+3] = (__bf16)f.w;
      }
      if (j0 + 32 < N_TOK)
        __builtin_prefetch(pbase + (size_t)(j0 + 32 + jr) * 128 + q * 32, 0, 1);
    }
    // A fragment: rows = heads (rows 8..15 zero)
    v16bf aa;
    const float* srow = S + (((size_t)b * H_C + (ln16 & 7)) * N_TOK + i) * N_TOK + j0;
    #pragma unroll
    for (int e = 0; e < 16; ++e)
      aa[e] = (ln16 < 8) ? (__bf16)srow[a_kidx(grp, e)] : (__bf16)0.f;
    __syncthreads();
    #pragma unroll
    for (int dt = 0; dt < 2; ++dt) {
      int d = (wave * 2 + dt) * 16 + ln16;
      v16bf bb;
      #pragma unroll
      for (int e = 0; e < 16; ++e) bb[e] = Ps[b_kidx(grp, e)][d];
      acc[dt] = wmma_bf16(aa, bb, acc[dt]);
    }
    __syncthreads();
  }
  if (grp == 0) {  // D rows r = head (grp1 rows are padding)
    #pragma unroll
    for (int dt = 0; dt < 2; ++dt) {
      int d = (wave * 2 + dt) * 16 + ln16;
      #pragma unroll
      for (int r = 0; r < 8; ++r)
        cat[((size_t)b * N_TOK + i) * CATW + 256 + r * 128 + d] = acc[dt][r];
    }
  }
}

// LayerNorm(out = LN(a + b) * g + beta).  One 128-thread block per row of 512.
__global__ void __launch_bounds__(128)
ln_kernel(const float* __restrict__ a, const float* __restrict__ bres,
          const float* __restrict__ g, const float* __restrict__ beta,
          float* __restrict__ out)
{
  __shared__ float red[128];
  size_t base = (size_t)blockIdx.x * DIM_C;
  int t = threadIdx.x;
  float v[4], s = 0.f;
  #pragma unroll
  for (int k = 0; k < 4; ++k) {
    v[k] = a[base + t + 128 * k] + bres[base + t + 128 * k];
    s += v[k];
  }
  float mean = block_reduce_sum(s, red) * (1.f / DIM_C);
  float vs = 0.f;
  #pragma unroll
  for (int k = 0; k < 4; ++k) { float d = v[k] - mean; vs += d * d; }
  float var = block_reduce_sum(vs, red) * (1.f / DIM_C);
  float inv = rsqrtf(var + LN_EPS_C);
  #pragma unroll
  for (int k = 0; k < 4; ++k) {
    int c = t + 128 * k;
    out[base + c] = (v[k] - mean) * inv * g[c] + beta[c];
  }
}

// ---------------------------------------------------------------------------
// Host launcher
// ---------------------------------------------------------------------------
extern "C" void kernel_launch(void* const* d_in, const int* in_sizes, int n_in,
                              void* d_out, int out_size, void* d_ws, size_t ws_size,
                              hipStream_t stream)
{
  (void)in_sizes; (void)n_in; (void)out_size; (void)ws_size;
  const float* x     = (const float*)d_in[0];
  const float* pair  = (const float*)d_in[1];
  const float* rot   = (const float*)d_in[2];
  const float* trans = (const float*)d_in[3];
  const float* Wq_s  = (const float*)d_in[4];
  const float* Wk_s  = (const float*)d_in[5];
  const float* Wv_s  = (const float*)d_in[6];
  const float* Wq_p  = (const float*)d_in[7];
  const float* Wk_p  = (const float*)d_in[8];
  const float* Wv_p  = (const float*)d_in[9];
  const float* pw    = (const float*)d_in[10];
  const float* Wpair = (const float*)d_in[11];
  const float* bpair = (const float*)d_in[12];
  const float* Wo    = (const float*)d_in[13];
  const float* bo    = (const float*)d_in[14];
  const float* ln1g  = (const float*)d_in[15];
  const float* ln1b  = (const float*)d_in[16];
  const float* ln2g  = (const float*)d_in[17];
  const float* ln2b  = (const float*)d_in[18];
  const float* ffw1  = (const float*)d_in[19];
  const float* ffb1  = (const float*)d_in[20];
  const float* ffw2  = (const float*)d_in[21];
  const float* ffb2  = (const float*)d_in[22];
  const float* ffw3  = (const float*)d_in[23];
  const float* ffb3  = (const float*)d_in[24];

  float* W = (float*)d_ws;
  const size_t OFF_QS  = 0;
  const size_t OFF_KS  = OFF_QS  + 131072;
  const size_t OFF_VS  = OFF_KS  + 131072;
  const size_t OFF_QP  = OFF_VS  + 131072;
  const size_t OFF_KP  = OFF_QP  + 98304;
  const size_t OFF_VP  = OFF_KP  + 98304;
  const size_t OFF_Q2  = OFF_VP  + 98304;
  const size_t OFF_K2  = OFF_Q2  + 8192;
  const size_t OFF_S   = OFF_K2  + 8192;
  const size_t OFF_OPG = OFF_S   + 4194304;
  const size_t OFF_CAT = OFF_OPG + 98304;
  const size_t OFF_AO  = OFF_CAT + 1310720;
  const size_t OFF_A   = OFF_AO  + 524288;
  const size_t OFF_H1  = OFF_A   + 524288;
  const size_t OFF_H2  = OFF_H1  + 524288;
  const size_t OFF_F   = OFF_H2  + 524288;

  float* qs  = W + OFF_QS;  float* ks_ = W + OFF_KS;  float* vs  = W + OFF_VS;
  float* qp  = W + OFF_QP;  float* kp  = W + OFF_KP;  float* vp  = W + OFF_VP;
  float* q2w = W + OFF_Q2;  float* k2w = W + OFF_K2;  float* Sb  = W + OFF_S;
  float* opg = W + OFF_OPG; float* cat = W + OFF_CAT; float* ao  = W + OFF_AO;
  float* a   = W + OFF_A;   float* h1  = W + OFF_H1;  float* h2  = W + OFF_H2;
  float* f   = W + OFF_F;

  const int M = 2 * N_TOK;  // 1024

  // --- projections (WMMA GEMMs) ---
  gemm_bf16_kernel<<<dim3(16, 4), 128, 0, stream>>>(x, Wq_s, nullptr, qs,  M, 128, 512, 0);
  gemm_bf16_kernel<<<dim3(16, 4), 128, 0, stream>>>(x, Wk_s, nullptr, ks_, M, 128, 512, 0);
  gemm_bf16_kernel<<<dim3(16, 4), 128, 0, stream>>>(x, Wv_s, nullptr, vs,  M, 128, 512, 0);
  gemm_bf16_kernel<<<dim3(16, 3), 128, 0, stream>>>(x, Wq_p, nullptr, qp,  M,  96, 512, 0);
  gemm_bf16_kernel<<<dim3(16, 3), 128, 0, stream>>>(x, Wk_p, nullptr, kp,  M,  96, 512, 0);
  gemm_bf16_kernel<<<dim3(16, 3), 128, 0, stream>>>(x, Wv_p, nullptr, vp,  M,  96, 512, 0);

  // --- geometry ---
  point_global_kernel<<<384, 256, 0, stream>>>(qp, kp, vp, rot, trans);
  sq_kernel<<<64, 256, 0, stream>>>(qp, kp, pw, q2w, k2w);

  // --- logits + softmax ---
  pair_logits_kernel<<<8192, 128, 0, stream>>>(pair, Wpair, bpair, Sb);
  attn_logits_kernel<<<4096, 128, 0, stream>>>(qs, ks_, qp, kp, q2w, k2w, pw, Sb);
  softmax_kernel<<<8192, 128, 0, stream>>>(Sb);

  // --- attention outputs ---
  av_kernel<<<128, 128, 0, stream>>>(Sb, vs, vp, cat, opg);
  pl_kernel<<<128, 256, 0, stream>>>(opg, rot, trans, cat);
  out_pair_kernel<<<1024, 128, 0, stream>>>(Sb, pair, cat);

  // --- output projection + LN + FFN + LN ---
  gemm_bf16_kernel<<<dim3(16, 16), 128, 0, stream>>>(cat, Wo, bo, ao, M, 512, 1280, 0);
  ln_kernel<<<1024, 128, 0, stream>>>(ao, x, ln1g, ln1b, a);
  gemm_bf16_kernel<<<dim3(16, 16), 128, 0, stream>>>(a,  ffw1, ffb1, h1, M, 512, 512, 1);
  gemm_bf16_kernel<<<dim3(16, 16), 128, 0, stream>>>(h1, ffw2, ffb2, h2, M, 512, 512, 1);
  gemm_bf16_kernel<<<dim3(16, 16), 128, 0, stream>>>(h2, ffw3, ffb3, f,  M, 512, 512, 0);
  ln_kernel<<<1024, 128, 0, stream>>>(f, a, ln2g, ln2b, (float*)d_out);
}
